// Block_14328010899778
// MI455X (gfx1250) — compile-verified, matrix operand order checked by reading, not run on hardware
//
#include <hip/hip_runtime.h>
#include <hip/hip_bf16.h>

// ---------------------------------------------------------------------------
// RWKV-v5 block for MI455X (gfx1250, wave32, WMMA bf16 16x16x32,
// async Global->LDS double-buffered GEMM)
// ---------------------------------------------------------------------------

typedef __bf16 bf16_t;
typedef __attribute__((ext_vector_type(16))) __bf16 v16bf;
typedef __attribute__((ext_vector_type(8)))  float  v8f;

#define B_   8
#define TT_  2048
#define C_   2048
#define H_   32
#define S_   64
#define F_   7168
#define TCH  512            // recurrence chunk length
#define NCHUNK (TT_ / TCH)

static constexpr size_t NTOK = (size_t)B_ * TT_;   // 16384 token rows
static constexpr size_t XE   = NTOK * C_;          // elements of a [tokens, C] tensor
static constexpr size_t KE   = NTOK * F_;          // elements of a [tokens, F] tensor

__device__ __forceinline__ v8f v8f_zero() {
  v8f v;
#pragma unroll
  for (int i = 0; i < 8; ++i) v[i] = 0.f;
  return v;
}

__device__ __forceinline__ v8f wmma_bf16(v16bf a, v16bf b, v8f c) {
  // D = A(16x32 bf16) * B(32x16 bf16) + C(16x16 f32)
  return __builtin_amdgcn_wmma_f32_16x16x32_bf16(
      /*neg_a=*/false, a, /*neg_b=*/false, b,
      /*c_mod=*/(short)0, c, /*reuse_a=*/false, /*reuse_b=*/false);
}

// CDNA5 async Global->LDS copy, 16 bytes per lane, tracked by ASYNCcnt.
// vdst VGPR holds per-lane LDS byte address; vaddr pair holds global address.
__device__ __forceinline__ void async_copy_b128(unsigned lds_addr, const void* gaddr) {
  asm volatile("global_load_async_to_lds_b128 %0, %1, off"
               :: "v"(lds_addr), "v"(gaddr) : "memory");
}
__device__ __forceinline__ void wait_async0() {
  asm volatile("s_wait_asynccnt 0" ::: "memory");
}
__device__ __forceinline__ void wait_ds0() {
  asm volatile("s_wait_dscnt 0" ::: "memory");
}

// ---------------------------------------------------------------------------
// fp32 -> bf16 streaming conversion (weights), vectorized 4-wide
// ---------------------------------------------------------------------------
__global__ __launch_bounds__(256) void f2bf_kernel(const float* __restrict__ s,
                                                   bf16_t* __restrict__ d, size_t n4) {
  // n4 = n/4 (all weight sizes are multiples of 4)
  size_t i = (size_t)blockIdx.x * blockDim.x + threadIdx.x;
  size_t stride = (size_t)gridDim.x * blockDim.x;
  const float4* s4 = reinterpret_cast<const float4*>(s);
  uint2* d4 = reinterpret_cast<uint2*>(d);
  for (; i < n4; i += stride) {
    float4 v = s4[i];
    union { bf16_t b[4]; uint2 u; } t;
    t.b[0] = (bf16_t)v.x; t.b[1] = (bf16_t)v.y;
    t.b[2] = (bf16_t)v.z; t.b[3] = (bf16_t)v.w;
    d4[i] = t.u;
  }
}

// ---------------------------------------------------------------------------
// LayerNorm over C per token row; optionally emit last-token row (shift state)
// ---------------------------------------------------------------------------
__global__ __launch_bounds__(256) void ln_kernel(const float* __restrict__ x,
                                                 const float* __restrict__ g,
                                                 const float* __restrict__ b,
                                                 float* __restrict__ out,
                                                 float* __restrict__ last_out) {
  const int row = blockIdx.x;               // b*TT + t
  const int tid = threadIdx.x;
  const float* xr = x + (size_t)row * C_;
  float* orow = out + (size_t)row * C_;

  __shared__ float sbuf[256], s2buf[256];
  float s = 0.f, s2 = 0.f;
  for (int c = tid; c < C_; c += 256) { float v = xr[c]; s += v; s2 += v * v; }
  sbuf[tid] = s; s2buf[tid] = s2;
  __syncthreads();
#pragma unroll
  for (int off = 128; off > 0; off >>= 1) {
    if (tid < off) { sbuf[tid] += sbuf[tid + off]; s2buf[tid] += s2buf[tid + off]; }
    __syncthreads();
  }
  const float mean = sbuf[0] * (1.f / C_);
  const float var  = s2buf[0] * (1.f / C_) - mean * mean;
  const float inv  = rsqrtf(var + 1e-5f);
  const bool last  = (row % TT_) == (TT_ - 1);
  const int  bidx  = row / TT_;
  for (int c = tid; c < C_; c += 256) {
    float v = (xr[c] - mean) * inv * g[c] + b[c];
    orow[c] = v;
    if (last && last_out) last_out[(size_t)bidx * C_ + c] = v;
  }
}

// ---------------------------------------------------------------------------
// Token-shift mixes (time-mix: 3 outputs; channel-mix: 2 outputs), bf16 out
// ---------------------------------------------------------------------------
__global__ __launch_bounds__(256) void mix3_kernel(const float* __restrict__ xln,
                                                   const float* __restrict__ shift0,
                                                   const float* __restrict__ mk,
                                                   const float* __restrict__ mv,
                                                   const float* __restrict__ mr,
                                                   bf16_t* __restrict__ ok,
                                                   bf16_t* __restrict__ ov,
                                                   bf16_t* __restrict__ orr) {
  const int row = blockIdx.x;
  const int b = row / TT_, t = row % TT_;
  const float* cur  = xln + (size_t)row * C_;
  const float* prev = (t == 0) ? (shift0 + (size_t)b * C_) : (xln + (size_t)(row - 1) * C_);
  for (int c = threadIdx.x; c < C_; c += 256) {
    float xc = cur[c], xx = prev[c];
    size_t idx = (size_t)row * C_ + c;
    float k = mk[c], v = mv[c], r = mr[c];
    ok[idx]  = (bf16_t)(xc * k + xx * (1.f - k));
    ov[idx]  = (bf16_t)(xc * v + xx * (1.f - v));
    orr[idx] = (bf16_t)(xc * r + xx * (1.f - r));
  }
}

__global__ __launch_bounds__(256) void mix2_kernel(const float* __restrict__ xln,
                                                   const float* __restrict__ shift0,
                                                   const float* __restrict__ mk,
                                                   const float* __restrict__ mr,
                                                   bf16_t* __restrict__ ok,
                                                   bf16_t* __restrict__ orr) {
  const int row = blockIdx.x;
  const int b = row / TT_, t = row % TT_;
  const float* cur  = xln + (size_t)row * C_;
  const float* prev = (t == 0) ? (shift0 + (size_t)b * C_) : (xln + (size_t)(row - 1) * C_);
  for (int c = threadIdx.x; c < C_; c += 256) {
    float xc = cur[c], xx = prev[c];
    size_t idx = (size_t)row * C_ + c;
    float k = mk[c], r = mr[c];
    ok[idx]  = (bf16_t)(xc * k + xx * (1.f - k));
    orr[idx] = (bf16_t)(xc * r + xx * (1.f - r));
  }
}

// ---------------------------------------------------------------------------
// Tiled WMMA GEMM: out[m,n] = sum_k A[m,k] * W[n,k]   (A,W row-major bf16)
// Block = 256 threads (8 waves). Tile 128x128, K-step 32.
// Double-buffered LDS staging via async Global->LDS (ASYNCcnt);
// WMMA on buffer p overlaps the async stream into buffer p^1.
// Wave grid 2(M) x 4(N); each wave: 64x32 = 4x2 wmma 16x16 tiles.
// Epilogues: 0 store bf16; 1 relu^2 -> bf16; 2 res + acc -> f32;
//            3 acc -> f32; 4 outF += sigmoid(aux)*acc
// ---------------------------------------------------------------------------
template <int EPI>
__global__ __launch_bounds__(256) void gemm_bt_kernel(const bf16_t* __restrict__ A,
                                                      const bf16_t* __restrict__ Bw,
                                                      int M, int N, int K,
                                                      float* __restrict__ outF,
                                                      bf16_t* __restrict__ outB,
                                                      const float* __restrict__ aux) {
  __shared__ __align__(32) bf16_t As[2][128 * 32];
  __shared__ __align__(32) bf16_t Bs[2][128 * 32];

  const int tid  = threadIdx.x;
  const int wave = tid >> 5;
  const int lane = tid & 31;
  const int l15  = lane & 15;
  const int lhi  = lane >> 4;

  const int m0 = blockIdx.y * 128;
  const int n0 = blockIdx.x * 128;
  const int wm = (wave & 1) * 64;   // 2 waves along M
  const int wn = (wave >> 1) * 32;  // 4 waves along N

  v8f acc[4][2];
#pragma unroll
  for (int i = 0; i < 4; ++i)
#pragma unroll
    for (int j = 0; j < 2; ++j) acc[i][j] = v8f_zero();

  const int ldr = tid >> 1;           // 0..127 : tile row this thread stages
  const int ldc = (tid & 1) * 16;     // 0 or 16 : first of 16 halfs

  // issue the async copies for one K-tile into buffer `buf`
  auto issue_tile = [&](int buf, int k0) {
    const bf16_t* ga = A  + (size_t)(m0 + ldr) * K + k0 + ldc;
    const bf16_t* gb = Bw + (size_t)(n0 + ldr) * K + k0 + ldc;
    unsigned la = (unsigned)(uintptr_t)&As[buf][ldr * 32 + ldc];
    unsigned lb = (unsigned)(uintptr_t)&Bs[buf][ldr * 32 + ldc];
    async_copy_b128(la,      ga);
    async_copy_b128(la + 16, ga + 8);
    async_copy_b128(lb,      gb);
    async_copy_b128(lb + 16, gb + 8);
  };

  issue_tile(0, 0);
  int buf = 0;

  for (int k0 = 0; k0 < K; k0 += 32) {
    wait_async0();       // this wave's tile for `buf` has landed in LDS
    __syncthreads();     // every wave's tile landed; prev reads of buf^1 done

    if (k0 + 32 < K) issue_tile(buf ^ 1, k0 + 32);   // overlap next stream

    v16bf af[4], bfr[2];
#pragma unroll
    for (int tm = 0; tm < 4; ++tm)
      af[tm] = *reinterpret_cast<const v16bf*>(&As[buf][(wm + tm * 16 + l15) * 32 + lhi * 16]);
#pragma unroll
    for (int tn = 0; tn < 2; ++tn)
      bfr[tn] = *reinterpret_cast<const v16bf*>(&Bs[buf][(wn + tn * 16 + l15) * 32 + lhi * 16]);
#pragma unroll
    for (int tm = 0; tm < 4; ++tm)
#pragma unroll
      for (int tn = 0; tn < 2; ++tn)
        acc[tm][tn] = wmma_bf16(af[tm], bfr[tn], acc[tm][tn]);

    buf ^= 1;
  }

  // epilogue: C-layout: element e of v8f -> row = base + e + 8*lhi, col = nbase + l15
#pragma unroll
  for (int tm = 0; tm < 4; ++tm) {
#pragma unroll
    for (int tn = 0; tn < 2; ++tn) {
      const int rbase = m0 + wm + tm * 16 + lhi * 8;
      const int col   = n0 + wn + tn * 16 + l15;
#pragma unroll
      for (int e = 0; e < 8; ++e) {
        const size_t idx = (size_t)(rbase + e) * N + col;
        const float v = acc[tm][tn][e];
        if constexpr (EPI == 0) {
          outB[idx] = (bf16_t)v;
        } else if constexpr (EPI == 1) {
          float t = fmaxf(v, 0.f);
          outB[idx] = (bf16_t)(t * t);
        } else if constexpr (EPI == 2) {
          outF[idx] = aux[idx] + v;
        } else if constexpr (EPI == 3) {
          outF[idx] = v;
        } else {  // EPI == 4
          float sg = 1.f / (1.f + __expf(-aux[idx]));
          outF[idx] = outF[idx] + sg * v;
        }
      }
    }
  }
}

// ---------------------------------------------------------------------------
// Chunked WKV attention + GroupNorm. One workgroup per (b,h); 8 waves.
// State [64x64] f32 lives in LDS across the 4-chunk scan.
// ---------------------------------------------------------------------------
__global__ __launch_bounds__(256) void attn_kernel(const bf16_t* __restrict__ rb,
                                                   const bf16_t* __restrict__ kb,
                                                   const bf16_t* __restrict__ vb,
                                                   const float* __restrict__ s0,
                                                   const float* __restrict__ decay,
                                                   const float* __restrict__ faaaa,
                                                   const float* __restrict__ gn_g,
                                                   const float* __restrict__ gn_b,
                                                   bf16_t* __restrict__ yn,
                                                   float* __restrict__ wkv_out) {
  const int bh = blockIdx.x;
  const int b = bh / H_, h = bh % H_;
  const int tid = threadIdx.x, wave = tid >> 5, lane = tid & 31;
  const int l15 = lane & 15, lhi = lane >> 4;

  __shared__ __align__(32) float  sstate[S_ * S_];          // 16 KB
  __shared__ __align__(32) bf16_t sA[8][16 * 32];           //  8 KB score staging
  __shared__ __align__(32) float  yrow[8][16 * 64];         // 32 KB GN staging

  for (int i = tid; i < S_ * S_; i += 256) sstate[i] = s0[(size_t)bh * (S_ * S_) + i];
  __syncthreads();

  const float dec = __expf(-__expf(decay[h]));
  const float u   = faaaa[h];
  const float l2d = __log2f(dec);
  auto dpow = [l2d](float e) { return exp2f(l2d * e); };

  const size_t headoff = (size_t)h * S_;

  for (int ch = 0; ch < NCHUNK; ++ch) {
    const size_t base = (size_t)b * TT_ + (size_t)ch * TCH;  // first token row of chunk

    // ---------------- phase 1: y for all i-tiles (reads sstate) ------------
    for (int it = wave; it < TCH / 16; it += 8) {
      const int i0 = it * 16;
      // r A-fragments (16x32 each; K = S feature index)
      v16bf ra0 = *reinterpret_cast<const v16bf*>(rb + (base + i0 + l15) * C_ + headoff + lhi * 16);
      v16bf ra1 = *reinterpret_cast<const v16bf*>(rb + (base + i0 + l15) * C_ + headoff + 32 + lhi * 16);

      v8f y[4];
#pragma unroll
      for (int nt = 0; nt < 4; ++nt) y[nt] = v8f_zero();

      // carried state term: (r @ sstate) * wb[i]
#pragma unroll
      for (int nt = 0; nt < 4; ++nt) {
        v16bf b0, b1;
#pragma unroll
        for (int e = 0; e < 16; ++e) {
          b0[e] = (bf16_t)sstate[(lhi * 16 + e) * S_ + nt * 16 + l15];
          b1[e] = (bf16_t)sstate[(32 + lhi * 16 + e) * S_ + nt * 16 + l15];
        }
        v8f t = v8f_zero();
        t = wmma_bf16(ra0, b0, t);
        t = wmma_bf16(ra1, b1, t);
#pragma unroll
        for (int e = 0; e < 8; ++e) {
          float wb = dpow((float)(i0 + e + lhi * 8));
          y[nt][e] += t[e] * wb;
        }
      }

      // intra-chunk causal term
      for (int j0 = 0; j0 <= i0; j0 += 32) {
#pragma unroll
        for (int sub = 0; sub < 2; ++sub) {
          const int js = j0 + sub * 16;
          v8f sc = v8f_zero();
          if (js <= i0 + 15) {  // not fully masked
            v16bf kb0 = *reinterpret_cast<const v16bf*>(kb + (base + js + l15) * C_ + headoff + lhi * 16);
            v16bf kb1 = *reinterpret_cast<const v16bf*>(kb + (base + js + l15) * C_ + headoff + 32 + lhi * 16);
            sc = wmma_bf16(ra0, kb0, sc);
            sc = wmma_bf16(ra1, kb1, sc);
          }
#pragma unroll
          for (int e = 0; e < 8; ++e) {
            const int i = i0 + e + lhi * 8;
            const int j = js + l15;
            const int d = i - j;
            float w = (d > 0) ? dpow((float)(d - 1)) : ((d == 0) ? u : 0.f);
            sA[wave][(e + lhi * 8) * 32 + sub * 16 + l15] = (bf16_t)(sc[e] * w);
          }
        }
        wait_ds0();  // wave-local LDS RAW
        v16bf scA = *reinterpret_cast<const v16bf*>(&sA[wave][l15 * 32 + lhi * 16]);
#pragma unroll
        for (int nt = 0; nt < 4; ++nt) {
          v16bf vB;
#pragma unroll
          for (int e = 0; e < 16; ++e)
            vB[e] = vb[(base + j0 + lhi * 16 + e) * C_ + headoff + nt * 16 + l15];
          y[nt] = wmma_bf16(scA, vB, y[nt]);
        }
      }

      // GroupNorm over S per row, write yn (bf16)
#pragma unroll
      for (int nt = 0; nt < 4; ++nt)
#pragma unroll
        for (int e = 0; e < 8; ++e)
          yrow[wave][(e + lhi * 8) * 64 + nt * 16 + l15] = y[nt][e];
      wait_ds0();
      if (lane < 16) {
        const float* yr = &yrow[wave][lane * 64];
        float m = 0.f;
        for (int s = 0; s < 64; ++s) m += yr[s];
        m *= (1.f / 64.f);
        float var = 0.f;
        for (int s = 0; s < 64; ++s) { float d = yr[s] - m; var += d * d; }
        var *= (1.f / 64.f);
        const float inv = rsqrtf(var + 1e-5f);
        bf16_t* dst = yn + (base + i0 + lane) * C_ + headoff;
        for (int s = 0; s < 64; ++s)
          dst[s] = (bf16_t)((yr[s] - m) * inv * gn_g[headoff + s] + gn_b[headoff + s]);
      }
    }
    __syncthreads();

    // ---------------- phase 2: state update (writes sstate) ----------------
    const float wS = dpow((float)TCH);
    for (int tile = wave; tile < 16; tile += 8) {
      const int su0 = (tile >> 2) * 16;  // s (row of state)
      const int uu0 = (tile & 3) * 16;   // u (col of state)
      v8f acc = v8f_zero();
      for (int i = 0; i < TCH; i += 32) {
        v16bf aF, bF;
#pragma unroll
        for (int e = 0; e < 16; ++e) {
          const int ii = i + lhi * 16 + e;
          const float wk = dpow((float)(TCH - 1 - ii));
          const float kv = (float)kb[(base + ii) * C_ + headoff + su0 + l15];
          aF[e] = (bf16_t)(kv * wk);                                   // A[s,i] = k[i,s]*wk[i]
          bF[e] = vb[(base + ii) * C_ + headoff + uu0 + l15];          // B[i,u] = v[i,u]
        }
        acc = wmma_bf16(aF, bF, acc);
      }
#pragma unroll
      for (int e = 0; e < 8; ++e) {
        const int srow = su0 + e + lhi * 8;
        const int ucol = uu0 + l15;
        sstate[srow * S_ + ucol] = wS * sstate[srow * S_ + ucol] + acc[e];
      }
    }
    __syncthreads();
  }

  for (int i = tid; i < S_ * S_; i += 256)
    wkv_out[(size_t)bh * (S_ * S_) + i] = sstate[i];
}

// ---------------------------------------------------------------------------
// Host launcher
// ---------------------------------------------------------------------------
extern "C" void kernel_launch(void* const* d_in, const int* in_sizes, int n_in,
                              void* d_out, int out_size, void* d_ws, size_t ws_size,
                              hipStream_t stream) {
  (void)in_sizes; (void)n_in; (void)out_size; (void)ws_size;

  const float* x     = (const float*)d_in[0];
  const float* tmsh  = (const float*)d_in[1];
  const float* wkv0  = (const float*)d_in[2];
  const float* cmsh  = (const float*)d_in[3];
  const float* ln1g  = (const float*)d_in[4];
  const float* ln1b  = (const float*)d_in[5];
  const float* ln2g  = (const float*)d_in[6];
  const float* ln2b  = (const float*)d_in[7];
  const float* tmk   = (const float*)d_in[8];
  const float* tmv   = (const float*)d_in[9];
  const float* tmr   = (const float*)d_in[10];
  const float* decay = (const float*)d_in[11];
  const float* faaaa = (const float*)d_in[12];
  const float* Wr    = (const float*)d_in[13];
  const float* Wk    = (const float*)d_in[14];
  const float* Wv    = (const float*)d_in[15];
  const float* Wo    = (const float*)d_in[16];
  const float* gng   = (const float*)d_in[17];
  const float* gnb   = (const float*)d_in[18];
  const float* cmk   = (const float*)d_in[19];
  const float* cmr   = (const float*)d_in[20];
  const float* Wkf   = (const float*)d_in[21];
  const float* Wvf   = (const float*)d_in[22];
  const float* Wrf   = (const float*)d_in[23];

  float* out_x   = (float*)d_out;                         // [B,TT,C]
  float* out_tm  = out_x + XE;                            // [B,C]
  float* out_wkv = out_tm + (size_t)B_ * C_;              // [B,H,S,S]
  float* out_cm  = out_wkv + (size_t)B_ * H_ * S_ * S_;   // [B,C]

  // ---- workspace carve-up (256B aligned) ----
  char* p = (char*)d_ws;
  auto take = [&p](size_t bytes) { char* r = p; p += (bytes + 255) & ~(size_t)255; return r; };
  float*  xln  = (float*)take(XE * 4);
  bf16_t* mxk  = (bf16_t*)take(XE * 2);
  bf16_t* mxv  = (bf16_t*)take(XE * 2);
  bf16_t* mxr  = (bf16_t*)take(XE * 2);
  bf16_t* rB   = (bf16_t*)take(XE * 2);
  bf16_t* kB   = (bf16_t*)take(XE * 2);
  bf16_t* vB   = (bf16_t*)take(XE * 2);
  bf16_t* ynB  = (bf16_t*)take(XE * 2);
  float*  rr   = (float*)take(XE * 4);
  bf16_t* kk   = (bf16_t*)take(KE * 2);
  bf16_t* wr_b  = (bf16_t*)take((size_t)C_ * C_ * 2);
  bf16_t* wk_b  = (bf16_t*)take((size_t)C_ * C_ * 2);
  bf16_t* wv_b  = (bf16_t*)take((size_t)C_ * C_ * 2);
  bf16_t* wo_b  = (bf16_t*)take((size_t)C_ * C_ * 2);
  bf16_t* wrf_b = (bf16_t*)take((size_t)C_ * C_ * 2);
  bf16_t* wkf_b = (bf16_t*)take((size_t)F_ * C_ * 2);
  bf16_t* wvf_b = (bf16_t*)take((size_t)C_ * F_ * 2);

  // ---- weight conversion fp32 -> bf16 (vectorized) ----
  const int CVB = 2048;
  f2bf_kernel<<<CVB, 256, 0, stream>>>(Wr,  wr_b,  (size_t)C_ * C_ / 4);
  f2bf_kernel<<<CVB, 256, 0, stream>>>(Wk,  wk_b,  (size_t)C_ * C_ / 4);
  f2bf_kernel<<<CVB, 256, 0, stream>>>(Wv,  wv_b,  (size_t)C_ * C_ / 4);
  f2bf_kernel<<<CVB, 256, 0, stream>>>(Wo,  wo_b,  (size_t)C_ * C_ / 4);
  f2bf_kernel<<<CVB, 256, 0, stream>>>(Wrf, wrf_b, (size_t)C_ * C_ / 4);
  f2bf_kernel<<<CVB, 256, 0, stream>>>(Wkf, wkf_b, (size_t)F_ * C_ / 4);
  f2bf_kernel<<<CVB, 256, 0, stream>>>(Wvf, wvf_b, (size_t)C_ * F_ / 4);

  const int M = (int)NTOK;  // 16384

  // ---- LN1 + time-mix token shift ----
  ln_kernel<<<M, 256, 0, stream>>>(x, ln1g, ln1b, xln, out_tm);
  mix3_kernel<<<M, 256, 0, stream>>>(xln, tmsh, tmk, tmv, tmr, mxk, mxv, mxr);

  // ---- r/k/v projections (WMMA bf16, async double-buffered) ----
  dim3 gC(C_ / 128, M / 128);
  gemm_bt_kernel<0><<<gC, 256, 0, stream>>>(mxr, wr_b, M, C_, C_, nullptr, rB, nullptr);
  gemm_bt_kernel<0><<<gC, 256, 0, stream>>>(mxk, wk_b, M, C_, C_, nullptr, kB, nullptr);
  gemm_bt_kernel<0><<<gC, 256, 0, stream>>>(mxv, wv_b, M, C_, C_, nullptr, vB, nullptr);

  // ---- chunked WKV attention + GroupNorm ----
  attn_kernel<<<B_ * H_, 256, 0, stream>>>(rB, kB, vB, wkv0, decay, faaaa,
                                           gng, gnb, ynB, out_wkv);

  // ---- output projection + residual: x1 = x + yn @ Wo^T ----
  gemm_bt_kernel<2><<<gC, 256, 0, stream>>>(ynB, wo_b, M, C_, C_, out_x, nullptr, x);

  // ---- LN2 + channel-mix token shift ----
  ln_kernel<<<M, 256, 0, stream>>>(out_x, ln2g, ln2b, xln, out_cm);
  mix2_kernel<<<M, 256, 0, stream>>>(xln, cmsh, cmk, cmr, mxk, mxr);

  // ---- channel mix: kk = relu(xk @ Wkf^T)^2 ; rr = xr @ Wrf^T ;
  //      x += sigmoid(rr) * (kk @ Wvf^T) ----
  dim3 gF(F_ / 128, M / 128);
  gemm_bt_kernel<1><<<gF, 256, 0, stream>>>(mxk, wkf_b, M, F_, C_, nullptr, kk, nullptr);
  gemm_bt_kernel<3><<<gC, 256, 0, stream>>>(mxr, wrf_b, M, C_, C_, rr, nullptr, nullptr);
  gemm_bt_kernel<4><<<gC, 256, 0, stream>>>(kk, wvf_b, M, C_, F_, out_x, nullptr, rr);
}
